// MultiHeadAttention_64914135712370
// MI455X (gfx1250) — compile-verified, matrix operand order checked by reading, not run on hardware
//
#include <hip/hip_runtime.h>
#include <hip/hip_bf16.h>

// ---------------------------------------------------------------------------
// MHA forward for MI455X (gfx1250, wave32, WMMA).
//   D_MODEL=1024, H=16, DK=64, B=2, S=2048  -> 4096 tokens.
// Pipeline: fp32->bf16 casts, QKV GEMM (WMMA bf16 + async A-tile staging into
// LDS, double buffered), flash attention (WMMA bf16, online softmax, P staged
// through LDS), output GEMM (same WMMA tile, fp32 out).
// ---------------------------------------------------------------------------

typedef __attribute__((ext_vector_type(16))) __bf16 v16bf;
typedef __attribute__((ext_vector_type(8)))  float  v8f;
typedef int v4i_vec __attribute__((vector_size(16)));
typedef __attribute__((address_space(1))) v4i_vec* gptr_b128;
typedef __attribute__((address_space(3))) v4i_vec* lptr_b128;

#define NTOK   4096   // b*s tokens
#define DMODEL 1024
#define NHEADS 16
#define DK     64
#define SEQ    2048
#define NQKV   3072

#if defined(__has_builtin)
#if __has_builtin(__builtin_amdgcn_global_load_async_to_lds_b128)
#define HAVE_ASYNC_LDS 1
#endif
#endif
#ifndef HAVE_ASYNC_LDS
#define HAVE_ASYNC_LDS 0
#endif

__device__ __forceinline__ unsigned short f32_to_bf16(float f) {
    union { float f; unsigned int u; } v; v.f = f;
    unsigned int rnd = 0x7FFFu + ((v.u >> 16) & 1u);   // round-to-nearest-even
    return (unsigned short)((v.u + rnd) >> 16);
}

__device__ __forceinline__ v16bf ldfrag(const unsigned short* p) {
    return *(const v16bf*)p;   // 32B, callers guarantee 32B alignment
}

__device__ __forceinline__ v8f wmma_bf16(v16bf a, v16bf b, v8f c) {
    // (neg_a, A, neg_b, B, c_mod, C, reuse_a, reuse_b)
    return __builtin_amdgcn_wmma_f32_16x16x32_bf16(false, a, false, b,
                                                   (short)0, c, false, false);
}

// --- 16B global -> LDS copy: async on CDNA5 (ASYNCcnt), sync fallback -------
__device__ __forceinline__ void cp16_g2l(const unsigned short* g, unsigned short* l) {
#if HAVE_ASYNC_LDS
    __builtin_amdgcn_global_load_async_to_lds_b128(
        (gptr_b128)(__attribute__((address_space(1))) void*)g,
        (lptr_b128)(__attribute__((address_space(3))) void*)l,
        0, 0);
#else
    *(uint4*)l = *(const uint4*)g;
#endif
}

template <int N>
__device__ __forceinline__ void wait_async_le() {
#if HAVE_ASYNC_LDS
#if __has_builtin(__builtin_amdgcn_s_wait_asynccnt)
    __builtin_amdgcn_s_wait_asynccnt(N);
#else
    asm volatile("s_wait_asynccnt %0" :: "i"(N) : "memory");
#endif
#endif
}

// ---------------------------------------------------------------------------
// fp32 -> bf16 elementwise cast
// ---------------------------------------------------------------------------
__global__ void cast_f32_bf16_kernel(const float* __restrict__ in,
                                     unsigned short* __restrict__ out, int n) {
    int i = blockIdx.x * blockDim.x + threadIdx.x;
    if (i < n) out[i] = f32_to_bf16(in[i]);
}

// ---------------------------------------------------------------------------
// WMMA GEMM: C[M,N] = A[M,K] * B[N,K]^T, A/B bf16 K-major, fp32 accumulate.
// Block = 4 waves sharing a 64-row A slab staged (async) into LDS, double
// buffered across K=32 steps; each wave owns a 64x64 tile (16 WMMAs / step).
// EPI==0: QKV epilogue -> scatter bf16 into Q[b,h,s,dk], K[b,h,s,dk], Vt[b,h,dk,s]
// EPI==1: fp32 epilogue -> Co[M,N]
// ---------------------------------------------------------------------------
template <int EPI>
__global__ __launch_bounds__(128)
void gemm_bf16_kernel(const unsigned short* __restrict__ A,
                      const unsigned short* __restrict__ B,
                      int M, int N, int Kdim,
                      unsigned short* __restrict__ Qo,
                      unsigned short* __restrict__ Ko,
                      unsigned short* __restrict__ Vto,
                      float* __restrict__ Co) {
    __shared__ __align__(32) unsigned short Atile[2][64 * 32];

    const int tid  = threadIdx.x;
    const int lane = tid & 31;
    const int wave = tid >> 5;
    const int m0 = blockIdx.x * 64;
    const int n0 = (blockIdx.y * 4 + wave) * 64;
    const int lrow = lane & 15;           // M (A) / N (B) index within tile
    const int lhi  = (lane >> 4) * 16;    // K sub-offset (elements)
    const int mhi  = (lane >> 4) * 8;     // C-row offset for high half-wave

    // staging map: 128 threads x 32B cover the 64x32 bf16 slab (4KB)
    const int srow = tid >> 1;            // 0..63
    const int scol = (tid & 1) * 16;      // element offset 0 / 16
    const unsigned short* Ag = A + (size_t)(m0 + srow) * Kdim + scol;

    v8f acc[4][4];
#pragma unroll
    for (int i = 0; i < 4; ++i)
#pragma unroll
        for (int j = 0; j < 4; ++j) acc[i][j] = v8f{0,0,0,0,0,0,0,0};

    const int nk = Kdim / 32;

    // prologue: stage K-slab 0 into buffer 0 (2 x b128 per thread)
    {
        unsigned short* dst = &Atile[0][srow * 32 + scol];
        cp16_g2l(Ag + 0, dst);
        cp16_g2l(Ag + 8, dst + 8);
    }

    for (int kt = 0; kt < nk; ++kt) {
        const int cur = kt & 1;
        const int k0  = kt * 32;
        if (kt + 1 < nk) {   // overlap next slab copy with this step's WMMAs
            unsigned short* dst = &Atile[1 - cur][srow * 32 + scol];
            cp16_g2l(Ag + k0 + 32, dst);
            cp16_g2l(Ag + k0 + 40, dst + 8);
            __builtin_prefetch(B + (size_t)(n0 + lrow) * Kdim + k0 + 32, 0, 1);
            wait_async_le<2>();   // retire current slab; next (2 ops) in flight
        } else {
            wait_async_le<0>();
        }
        __syncthreads();          // slab visible to all 4 waves

        v16bf af[4], bfr[4];
#pragma unroll
        for (int i = 0; i < 4; ++i)
            af[i] = ldfrag(&Atile[cur][(i * 16 + lrow) * 32 + lhi]);
#pragma unroll
        for (int j = 0; j < 4; ++j)
            bfr[j] = ldfrag(B + (size_t)(n0 + j * 16 + lrow) * Kdim + k0 + lhi);
#pragma unroll
        for (int i = 0; i < 4; ++i)
#pragma unroll
            for (int j = 0; j < 4; ++j)
                acc[i][j] = wmma_bf16(af[i], bfr[j], acc[i][j]);

        __syncthreads();          // readers done before this buffer is reused
    }

#pragma unroll
    for (int i = 0; i < 4; ++i)
#pragma unroll
        for (int j = 0; j < 4; ++j)
#pragma unroll
            for (int e = 0; e < 8; ++e) {
                int m = m0 + i * 16 + e + mhi;          // token row
                int n = n0 + j * 16 + lrow;             // output column
                float v = acc[i][j][e];
                if (EPI == 0) {
                    // n -> (which, head, dk);  m -> (batch, s)
                    int three = n >> 10;
                    int h     = (n >> 6) & 15;
                    int d     = n & 63;
                    int b     = m >> 11;
                    int s     = m & 2047;
                    size_t bh = (size_t)b * NHEADS + h;
                    unsigned short bv = f32_to_bf16(v);
                    if (three == 0)      Qo [(bh * SEQ + s) * DK + d] = bv;
                    else if (three == 1) Ko [(bh * SEQ + s) * DK + d] = bv;
                    else                 Vto[(bh * DK + d) * SEQ + s] = bv;
                } else {
                    Co[(size_t)m * N + n] = v;
                }
            }
}

// ---------------------------------------------------------------------------
// Flash attention: one wave owns a 16-row q-tile of one (b,h).
// k processed in 32-wide tiles: S = Q*K^T (4 WMMAs), causal mask, online
// softmax, P -> LDS (bf16) -> A-operand, O += P*Vt (4 WMMAs).
// ---------------------------------------------------------------------------
__global__ __launch_bounds__(128)
void flash_attn_kernel(const unsigned short* __restrict__ Q,
                       const unsigned short* __restrict__ Kb,
                       const unsigned short* __restrict__ Vt,
                       unsigned short* __restrict__ AO) {
    __shared__ __align__(32) unsigned short Plds[4][16 * 32];

    const int lane = threadIdx.x & 31;
    const int wave = threadIdx.x >> 5;
    const int qt   = blockIdx.x * 4 + wave;        // global q-tile id
    const int qtiles = SEQ / 16;                   // 128
    const int bh = qt / qtiles;                    // 0..31
    const int q0 = (qt % qtiles) * 16;
    const int b  = bh >> 4, h = bh & 15;

    const unsigned short* Qh = Q  + (size_t)bh * SEQ * DK;
    const unsigned short* Kh = Kb + (size_t)bh * SEQ * DK;
    const unsigned short* Vh = Vt + (size_t)bh * DK * SEQ;
    unsigned short* Pw = Plds[wave];

    const int lrow = lane & 15;
    const int lhi  = (lane >> 4) * 16;
    const int mhi  = (lane >> 4) * 8;

    // Q fragments for the whole dk=64 (two K=32 slabs)
    v16bf aq[2];
#pragma unroll
    for (int j = 0; j < 2; ++j)
        aq[j] = ldfrag(Qh + (size_t)(q0 + lrow) * DK + j * 32 + lhi);

    float mrow[8], lsum[8];
    v8f o[4];
#pragma unroll
    for (int e = 0; e < 8; ++e) { mrow[e] = -1e30f; lsum[e] = 0.f; }
#pragma unroll
    for (int t = 0; t < 4; ++t) o[t] = v8f{0,0,0,0,0,0,0,0};

    const float sc = 0.125f;                       // 1/sqrt(64)

    for (int k0 = 0; k0 < q0 + 16; k0 += 32) {     // causal upper bound
        // ---- scores: two 16x16 subtiles covering k0..k0+31 ----
        v8f s[2];
#pragma unroll
        for (int n = 0; n < 2; ++n) {
            s[n] = v8f{0,0,0,0,0,0,0,0};
            int kc = k0 + n * 16;
#pragma unroll
            for (int j = 0; j < 2; ++j) {
                v16bf bk = ldfrag(Kh + (size_t)(kc + lrow) * DK + j * 32 + lhi);
                s[n] = wmma_bf16(aq[j], bk, s[n]);
            }
        }
        // ---- scale + causal mask ----
#pragma unroll
        for (int n = 0; n < 2; ++n)
#pragma unroll
            for (int e = 0; e < 8; ++e) {
                int qrow = q0 + e + mhi;
                int kcol = k0 + n * 16 + lrow;
                float v = s[n][e] * sc;
                s[n][e] = (kcol <= qrow) ? v : -1e30f;
            }
        // ---- online softmax per row, rescale O, stage P in LDS ----
#pragma unroll
        for (int e = 0; e < 8; ++e) {
            float rm = fmaxf(s[0][e], s[1][e]);
#pragma unroll
            for (int off = 8; off; off >>= 1)
                rm = fmaxf(rm, __shfl_xor(rm, off, 32));
            float mnew  = fmaxf(mrow[e], rm);
            float alpha = __expf(mrow[e] - mnew);
            float p0 = __expf(s[0][e] - mnew);
            float p1 = __expf(s[1][e] - mnew);
            float rs = p0 + p1;
#pragma unroll
            for (int off = 8; off; off >>= 1)
                rs += __shfl_xor(rs, off, 32);
            lsum[e] = lsum[e] * alpha + rs;
            mrow[e] = mnew;
#pragma unroll
            for (int t = 0; t < 4; ++t) o[t][e] *= alpha;
            int prow = e + mhi;
            Pw[prow * 32 + lrow]      = f32_to_bf16(p0);
            Pw[prow * 32 + 16 + lrow] = f32_to_bf16(p1);
        }
        asm volatile("" ::: "memory");   // keep LDS write->read ordered
        // ---- O += P * V ----
        v16bf ap = ldfrag(Pw + lrow * 32 + lhi);
#pragma unroll
        for (int t = 0; t < 4; ++t) {
            v16bf bv = ldfrag(Vh + (size_t)(t * 16 + lrow) * SEQ + k0 + lhi);
            o[t] = wmma_bf16(ap, bv, o[t]);
        }
    }

    // ---- normalize and store attn output as bf16 tokens x d_model ----
#pragma unroll
    for (int t = 0; t < 4; ++t)
#pragma unroll
        for (int e = 0; e < 8; ++e) {
            int tok = b * SEQ + q0 + e + mhi;
            int col = h * DK + t * 16 + lrow;
            AO[(size_t)tok * DMODEL + col] = f32_to_bf16(o[t][e] / lsum[e]);
        }
}

// ---------------------------------------------------------------------------
// Host-side orchestration
// ---------------------------------------------------------------------------
extern "C" void kernel_launch(void* const* d_in, const int* in_sizes, int n_in,
                              void* d_out, int out_size, void* d_ws, size_t ws_size,
                              hipStream_t stream) {
    const float* x     = (const float*)d_in[0];   // [2,2048,1024]
    const float* Wqkv  = (const float*)d_in[1];   // [3072,1024]
    const float* Wout  = (const float*)d_in[2];   // [1024,1024]
    float* out = (float*)d_out;                   // [2,2048,1024]

    char* ws = (char*)d_ws;
    size_t off = 0;
    auto carve = [&](size_t bytes) -> void* {
        void* p = ws + off;
        off += (bytes + 255) & ~(size_t)255;
        return p;
    };
    unsigned short* xb   = (unsigned short*)carve((size_t)NTOK * DMODEL * 2);
    unsigned short* wqb  = (unsigned short*)carve((size_t)NQKV * DMODEL * 2);
    unsigned short* wob  = (unsigned short*)carve((size_t)DMODEL * DMODEL * 2);
    unsigned short* Qb   = (unsigned short*)carve((size_t)NTOK * DMODEL * 2);
    unsigned short* Kbuf = (unsigned short*)carve((size_t)NTOK * DMODEL * 2);
    unsigned short* Vtb  = (unsigned short*)carve((size_t)NTOK * DMODEL * 2);
    unsigned short* aob  = (unsigned short*)carve((size_t)NTOK * DMODEL * 2);

    // 1) casts to bf16
    {
        int n;
        n = NTOK * DMODEL;
        cast_f32_bf16_kernel<<<(n + 255) / 256, 256, 0, stream>>>(x, xb, n);
        n = NQKV * DMODEL;
        cast_f32_bf16_kernel<<<(n + 255) / 256, 256, 0, stream>>>(Wqkv, wqb, n);
        n = DMODEL * DMODEL;
        cast_f32_bf16_kernel<<<(n + 255) / 256, 256, 0, stream>>>(Wout, wob, n);
    }

    // 2) QKV projection: [4096,1024] x [3072,1024]^T, scatter to Q/K/Vt
    {
        dim3 grid(NTOK / 64, NQKV / 64 / 4);  // (64, 12)
        gemm_bf16_kernel<0><<<grid, 128, 0, stream>>>(
            xb, wqb, NTOK, NQKV, DMODEL, Qb, Kbuf, Vtb, nullptr);
    }

    // 3) flash attention: 32 (b,h) * 128 q-tiles = 4096 waves, 4 per block
    flash_attn_kernel<<<(2 * NHEADS * (SEQ / 16)) / 4, 128, 0, stream>>>(
        Qb, Kbuf, Vtb, aob);

    // 4) output projection: [4096,1024] x [1024,1024]^T -> fp32 out
    {
        dim3 grid(NTOK / 64, DMODEL / 64 / 4); // (64, 4)
        gemm_bf16_kernel<1><<<grid, 128, 0, stream>>>(
            aob, wob, NTOK, DMODEL, DMODEL, nullptr, nullptr, nullptr, out);
    }
}
